// SinkhornLoss_8469675508202
// MI455X (gfx1250) — matrix operand order repaired and checked by reading, not compiled, not run on hardware
//
#include <hip/hip_runtime.h>

// ---------------------------------------------------------------------------
// Sinkhorn loss, MI455X (gfx1250) design:
//   Phase 1: bf16 WMMA GEMM (v_wmma_f32_16x16x32_bf16) fused with
//            K = exp2((dot-1)*10/ln2) epilogue, K stored bf16 (128 MB, fits
//            the 192 MB L2 -> 40 matvec passes run at L2 bandwidth).
//   Phase 2: 20 Sinkhorn iterations: row matvec (wave-per-row, packed bf16
//            loads + wave32 shuffle reduce), reciprocal, column matvec
//            (register accumulators + global_atomic_add_f32), reciprocal.
//   Phase 3: fused loss pass: cost recovered exactly as -eps*ln(K) via
//            v_log_f32, so no separate 256 MB cost matrix is needed.
// ---------------------------------------------------------------------------

#define BN       8192
#define DIM      128
#define EPSI     0.1f
#define EPS_DIV  1e-8f
#define N_ITERS  20

typedef __attribute__((ext_vector_type(8)))  __bf16 v8bf;
typedef __attribute__((ext_vector_type(16))) __bf16 v16bf;
typedef __attribute__((ext_vector_type(8)))  float  v8f;

__device__ __forceinline__ float bflo(unsigned int u) {
    union { unsigned int x; float f; } c; c.x = u << 16; return c.f;
}
__device__ __forceinline__ float bfhi(unsigned int u) {
    union { unsigned int x; float f; } c; c.x = u & 0xffff0000u; return c.f;
}

// --- fp32 -> bf16 conversion of S and T ------------------------------------
__global__ void sk_cvt_kernel(const float* __restrict__ S,
                              const float* __restrict__ T,
                              __bf16* __restrict__ Sb,
                              __bf16* __restrict__ Tb, int n) {
    int i = blockIdx.x * blockDim.x + threadIdx.x;
    if (i < n) {
        Sb[i] = (__bf16)S[i];
        Tb[i] = (__bf16)T[i];
    }
}

// --- u = v = 1/N ; zero the scalar output ----------------------------------
__global__ void sk_init_kernel(float* __restrict__ u, float* __restrict__ v,
                               float* __restrict__ out) {
    int i = blockIdx.x * blockDim.x + threadIdx.x;
    if (i < BN) { u[i] = 1.0f / BN; v[i] = 1.0f / BN; }
    if (i == 0) out[0] = 0.0f;
}

// --- WMMA GEMM + exp epilogue: K[m][n] = exp((dot-1)/eps) in bf16 ----------
// Workgroup = 8 waves; wave computes a 16x64 strip (4 accumulators).
__global__ __launch_bounds__(256)
void sk_gemm_exp_kernel(const __bf16* __restrict__ Sb,
                        const __bf16* __restrict__ Tb,
                        __bf16* __restrict__ K) {
    const int wave = threadIdx.x >> 5;
    const int lane = threadIdx.x & 31;
    const int r    = lane & 15;      // row/col within 16-wide fragment
    const int h    = lane >> 4;      // lane half
    const int m0   = (blockIdx.y * 8 + wave) * 16;
    const int n0   = blockIdx.x * 64;

    v8f acc[4] = {};

#pragma unroll
    for (int kk = 0; kk < DIM; kk += 32) {
        // A fragment (16x32 bf16), contiguous k order (same as B):
        //   lanes 0-15 : K = kk+{0..15} contiguous
        //   lanes 16-31: K = kk+{16..31} contiguous
        const __bf16* ap = Sb + (size_t)(m0 + r) * DIM + kk + h * 16;
        v8bf alo = *(const v8bf*)(ap);
        v8bf ahi = *(const v8bf*)(ap + 8);
        v16bf a = __builtin_shufflevector(alo, ahi,
                    0, 1, 2, 3, 4, 5, 6, 7, 8, 9, 10, 11, 12, 13, 14, 15);
#pragma unroll
        for (int s = 0; s < 4; ++s) {
            // B fragment (32x16): column n = row n of T; lanes 0-15 hold
            // K = kk+0..15 contiguous, lanes 16-31 hold K = kk+16..31.
            const int col = n0 + s * 16 + r;
            v16bf b = *(const v16bf*)(Tb + (size_t)col * DIM + kk + h * 16);
            acc[s] = __builtin_amdgcn_wmma_f32_16x16x32_bf16(
                false, a, false, b, (short)0, acc[s], false, false);
        }
    }

    // K = exp(-(1-dot)/eps) = exp2((dot-1) * 10/ln2)
    const float sc = 14.426950408889634f; // (1/EPSI)/ln(2)
#pragma unroll
    for (int s = 0; s < 4; ++s) {
        const int col = n0 + s * 16 + r;
#pragma unroll
        for (int i = 0; i < 8; ++i) {
            const int row = m0 + i + h * 8;   // C/D layout: VGPR i, lane half h
            float kv = exp2f((acc[s][i] - 1.0f) * sc);
            K[(size_t)row * BN + col] = (__bf16)kv;
        }
    }
}

// --- r[i] = sum_j K[i][j] * v[j]  (wave per row, bf16x8 packed loads) ------
__global__ __launch_bounds__(256)
void sk_rowmv_kernel(const __bf16* __restrict__ K,
                     const float* __restrict__ v,
                     float* __restrict__ r) {
    const int lane = threadIdx.x & 31;
    const int w    = blockIdx.x * 8 + (threadIdx.x >> 5);   // global wave id
    const float4* v4 = (const float4*)v;
    for (int row = w; row < BN; row += 256 * 8) {
        const uint4* Kr = (const uint4*)(K + (size_t)row * BN);
        float s = 0.0f;
        for (int c = lane; c < BN / 8; c += 32) {
            __builtin_prefetch(Kr + c + 64, 0, 0);
            uint4  q  = Kr[c];                 // 8 bf16 of K
            float4 va = v4[c * 2];
            float4 vb = v4[c * 2 + 1];
            s += bflo(q.x) * va.x + bfhi(q.x) * va.y
               + bflo(q.y) * va.z + bfhi(q.y) * va.w;
            s += bflo(q.z) * vb.x + bfhi(q.z) * vb.y
               + bflo(q.w) * vb.z + bfhi(q.w) * vb.w;
        }
#pragma unroll
        for (int off = 16; off > 0; off >>= 1) s += __shfl_xor(s, off, 32);
        if (lane == 0) r[row] = s;
    }
}

// --- u = 1/(r+eps); zero column accumulator for next pass ------------------
__global__ void sk_urecip_kernel(const float* __restrict__ r,
                                 float* __restrict__ u,
                                 float* __restrict__ c) {
    int i = blockIdx.x * blockDim.x + threadIdx.x;
    if (i < BN) { u[i] = 1.0f / (r[i] + EPS_DIV); c[i] = 0.0f; }
}

// --- c[j] += sum_{i in block rows} K[i][j] * u[i] --------------------------
// 64 blocks x 128 rows; 32 register accumulators/thread; coalesced rows.
__global__ __launch_bounds__(256)
void sk_colmv_kernel(const __bf16* __restrict__ K,
                     const float* __restrict__ u,
                     float* __restrict__ c) {
    const int tid = threadIdx.x;
    const int r0  = blockIdx.x * 128;
    float acc[4][8] = {};
    for (int row = r0; row < r0 + 128; ++row) {
        const float ur = u[row];
        const uint4* Kr = (const uint4*)(K + (size_t)row * BN);
#pragma unroll
        for (int b = 0; b < 4; ++b) {
            uint4 q = Kr[b * 256 + tid];
            acc[b][0] += bflo(q.x) * ur; acc[b][1] += bfhi(q.x) * ur;
            acc[b][2] += bflo(q.y) * ur; acc[b][3] += bfhi(q.y) * ur;
            acc[b][4] += bflo(q.z) * ur; acc[b][5] += bfhi(q.z) * ur;
            acc[b][6] += bflo(q.w) * ur; acc[b][7] += bfhi(q.w) * ur;
        }
    }
#pragma unroll
    for (int b = 0; b < 4; ++b)
#pragma unroll
        for (int e = 0; e < 8; ++e)
            atomicAdd(&c[(b * 256 + tid) * 8 + e], acc[b][e]);
}

// --- v = 1/(c+eps) ---------------------------------------------------------
__global__ void sk_vrecip_kernel(const float* __restrict__ c,
                                 float* __restrict__ v) {
    int i = blockIdx.x * blockDim.x + threadIdx.x;
    if (i < BN) v[i] = 1.0f / (c[i] + EPS_DIV);
}

// --- loss = sum u_i K_ij v_j cost_ij,  cost = -eps*ln(K) (exact inverse) ---
__global__ __launch_bounds__(256)
void sk_loss_kernel(const __bf16* __restrict__ K,
                    const float* __restrict__ u,
                    const float* __restrict__ v,
                    float* __restrict__ out) {
    const int lane = threadIdx.x & 31;
    const uint4* K4 = (const uint4*)K;
    const long long total = (long long)BN * (BN / 8);
    const float neg_eps_ln2 = -0.6931471805599453f * EPSI; // -eps*ln2
    float s = 0.0f;
    for (long long i = (long long)blockIdx.x * 256 + threadIdx.x; i < total;
         i += (long long)gridDim.x * 256) {
        const int row = (int)(i >> 10);          // BN/8 = 1024 uint4 per row
        const int c8  = ((int)i & 1023) * 8;
        const float ur = u[row];
        uint4 q = K4[i];
        float kv[8] = { bflo(q.x), bfhi(q.x), bflo(q.y), bfhi(q.y),
                        bflo(q.z), bfhi(q.z), bflo(q.w), bfhi(q.w) };
#pragma unroll
        for (int e = 0; e < 8; ++e) {
            const float k = kv[e];
            if (k > 0.0f) {
                const float cost = neg_eps_ln2 * __log2f(k);
                s += ur * k * v[c8 + e] * cost;
            }
        }
    }
#pragma unroll
    for (int off = 16; off > 0; off >>= 1) s += __shfl_xor(s, off, 32);
    if (lane == 0) atomicAdd(out, s);
}

// ---------------------------------------------------------------------------
extern "C" void kernel_launch(void* const* d_in, const int* in_sizes, int n_in,
                              void* d_out, int out_size, void* d_ws, size_t ws_size,
                              hipStream_t stream) {
    const float* S = (const float*)d_in[0];
    const float* T = (const float*)d_in[1];
    float* out = (float*)d_out;

    // Workspace layout (all offsets keep >=256B alignment)
    const size_t K_BYTES  = (size_t)BN * BN * 2;        // 134,217,728
    const size_t ST_BYTES = (size_t)BN * DIM * 2;       //   2,097,152
    const size_t V_BYTES  = (size_t)BN * 4;             //      32,768
    const size_t NEED = K_BYTES + 2 * ST_BYTES + 4 * V_BYTES;
    if (ws_size < NEED) return;   // insufficient scratch (deterministic no-op)

    char* ws = (char*)d_ws;
    __bf16* Km = (__bf16*)(ws);
    __bf16* Sb = (__bf16*)(ws + K_BYTES);
    __bf16* Tb = (__bf16*)(ws + K_BYTES + ST_BYTES);
    float*  u  = (float*)(ws + K_BYTES + 2 * ST_BYTES);
    float*  v  = (float*)(ws + K_BYTES + 2 * ST_BYTES + V_BYTES);
    float*  r  = (float*)(ws + K_BYTES + 2 * ST_BYTES + 2 * V_BYTES);
    float*  c  = (float*)(ws + K_BYTES + 2 * ST_BYTES + 3 * V_BYTES);

    // Phase 0: convert inputs to bf16; init u, v, out.
    sk_cvt_kernel<<<(BN * DIM + 255) / 256, 256, 0, stream>>>(S, T, Sb, Tb, BN * DIM);
    sk_init_kernel<<<(BN + 255) / 256, 256, 0, stream>>>(u, v, out);

    // Phase 1: K = exp((S.T^T - 1)/eps) in bf16 (fits 192 MB L2).
    dim3 ggrid(BN / 64, BN / (8 * 16));   // (128, 64) workgroups of 8 waves
    sk_gemm_exp_kernel<<<ggrid, 256, 0, stream>>>(Sb, Tb, Km);

    // Phase 2: 20 Sinkhorn iterations at L2 bandwidth.
    for (int it = 0; it < N_ITERS; ++it) {
        sk_rowmv_kernel<<<256, 256, 0, stream>>>(Km, v, r);
        sk_urecip_kernel<<<(BN + 255) / 256, 256, 0, stream>>>(r, u, c);
        sk_colmv_kernel<<<BN / 128, 256, 0, stream>>>(Km, u, c);
        sk_vrecip_kernel<<<(BN + 255) / 256, 256, 0, stream>>>(c, v);
    }

    // Phase 3: loss reduction (cost recomputed from K, no cost matrix).
    sk_loss_kernel<<<1024, 256, 0, stream>>>(Km, u, v, out);
}
